// IDM_SGC_52733608461009
// MI455X (gfx1250) — matrix operand order repaired and checked by reading, not compiled, not run on hardware
//
#include <hip/hip_runtime.h>
#include <math.h>

#define MDIM 64
#define NDIM 4096

typedef __attribute__((ext_vector_type(16))) __bf16 v16bf;
typedef __attribute__((ext_vector_type(8)))  float  v8f;

// ---------- bf16 helpers (round-to-nearest-even split) ----------
__device__ __forceinline__ unsigned short f2bf(float f) {
    unsigned int u = __float_as_uint(f);
    unsigned int r = (u + 0x7FFFu + ((u >> 16) & 1u)) >> 16;
    return (unsigned short)r;
}
__device__ __forceinline__ float bf2f(unsigned short h) {
    return __uint_as_float(((unsigned int)h) << 16);
}
__device__ __forceinline__ unsigned int pack2(unsigned short lo, unsigned short hi) {
    return ((unsigned int)hi << 16) | (unsigned int)lo;
}

// ============================================================================
// K1: G = F^T F / (||F^T F||_F + eps)   (64x64, one workgroup)
// ============================================================================
__global__ __launch_bounds__(256) void k_gram(const float* __restrict__ F,
                                              float* __restrict__ G) {
    __shared__ float Fl[64][64];
    __shared__ float FFl[64][64];
    __shared__ float red[256];
    int tid = threadIdx.x;
    for (int e = tid; e < 4096; e += 256) Fl[e >> 6][e & 63] = F[e];
    __syncthreads();
    float sq = 0.f;
    for (int e = tid; e < 4096; e += 256) {
        int i = e >> 6, j = e & 63;
        float s = 0.f;
        for (int k = 0; k < 64; ++k) s += Fl[k][i] * Fl[k][j];
        FFl[i][j] = s;
        sq += s * s;
    }
    red[tid] = sq;
    __syncthreads();
    for (int s = 128; s > 0; s >>= 1) {
        if (tid < s) red[tid] += red[tid + s];
        __syncthreads();
    }
    float inv = 1.f / (sqrtf(red[0]) + 1e-12f);
    for (int e = tid; e < 4096; e += 256) G[e] = FFl[e >> 6][e & 63] * inv;
}

// ============================================================================
// K2: Gp[k] = G^(2^k), k = 1..6  (Gp[0] = G already present). One workgroup.
// ============================================================================
__global__ __launch_bounds__(256) void k_gpow(float* __restrict__ Gp) {
    __shared__ float Ab[2][64][64];
    int tid = threadIdx.x;
    for (int e = tid; e < 4096; e += 256) Ab[0][e >> 6][e & 63] = Gp[e];
    __syncthreads();
    int cur = 0;
    for (int k = 1; k <= 6; ++k) {
        for (int e = tid; e < 4096; e += 256) {
            int i = e >> 6, j = e & 63;
            float s = 0.f;
            for (int t = 0; t < 64; ++t) s += Ab[cur][i][t] * Ab[cur][t][j];
            Ab[1 - cur][i][j] = s;
            Gp[k * 4096 + e] = s;
        }
        __syncthreads();
        cur = 1 - cur;
    }
}

// ============================================================================
// K3/K5: Out(64xN) = A(64xN) @ (TRANSB ? B^T : B), B = Q_S (NxN), split-bf16
// WMMA. 4 waves per WG, each wave owns a 16-column stripe, full M=64.
// A tile (64x32) staged to LDS as hi/lo bf16, row stride 18 dwords
// (conflict-free for the fragment gather pattern).
// ============================================================================
template <bool TRANSB>
__global__ __launch_bounds__(128) void k_gemm_wmma(const float* __restrict__ A,
                                                   const float* __restrict__ B,
                                                   float* __restrict__ Out) {
    __shared__ unsigned short Ahi[64 * 36];
    __shared__ unsigned short Alo[64 * 36];
    unsigned int* AHd = (unsigned int*)Ahi;
    unsigned int* ALd = (unsigned int*)Alo;

    const int tid  = threadIdx.x;
    const int lane = tid & 31;
    const int wave = tid >> 5;
    const int n0   = blockIdx.x * 64 + wave * 16;
    const int coln = n0 + (lane & 15);
    const int kb   = (lane >> 4) * 16;  // B-fragment K base (contiguous K, 16 per lane half, matches A)
    const int ak2  = (lane >> 4) * 8;   // A-fragment dword base (contiguous K, matches B)

    v8f acc[4];
#pragma unroll
    for (int mt = 0; mt < 4; ++mt)
        acc[mt] = (v8f){0.f, 0.f, 0.f, 0.f, 0.f, 0.f, 0.f, 0.f};

    for (int k0 = 0; k0 < NDIM; k0 += 32) {
        __syncthreads();
        // ---- stage A tile 64x32 as split bf16 (coalesced float4 loads) ----
#pragma unroll
        for (int v = 0; v < 4; ++v) {
            int fi = v * 128 + tid;
            int row = fi >> 3, c4 = fi & 7;
            const float4 f = *(const float4*)(A + row * NDIM + k0 + c4 * 4);
            unsigned short h0 = f2bf(f.x), h1 = f2bf(f.y), h2 = f2bf(f.z), h3 = f2bf(f.w);
            unsigned short l0 = f2bf(f.x - bf2f(h0)), l1 = f2bf(f.y - bf2f(h1));
            unsigned short l2 = f2bf(f.z - bf2f(h2)), l3 = f2bf(f.w - bf2f(h3));
            int base = row * 18 + c4 * 2;
            AHd[base]     = pack2(h0, h1);
            AHd[base + 1] = pack2(h2, h3);
            ALd[base]     = pack2(l0, l1);
            ALd[base + 1] = pack2(l2, l3);
        }
        __syncthreads();

        // ---- B fragment (32x16), split bf16, direct from global ----
        union { v16bf v; unsigned short s[16]; } bh, bl;
        if (TRANSB) {
            // B[k][n] = Q_S[n][k]: each lane streams 64B of a Q_S row
            const float4* bp = (const float4*)(B + (size_t)coln * NDIM + k0 + kb);
            float4 q0 = bp[0], q1 = bp[1], q2 = bp[2], q3 = bp[3];
            float bv[16] = {q0.x, q0.y, q0.z, q0.w, q1.x, q1.y, q1.z, q1.w,
                            q2.x, q2.y, q2.z, q2.w, q3.x, q3.y, q3.z, q3.w};
#pragma unroll
            for (int h = 0; h < 16; ++h) {
                bh.s[h] = f2bf(bv[h]);
                bl.s[h] = f2bf(bv[h] - bf2f(bh.s[h]));
            }
        } else {
            // B[k][n] = Q_S[k][n]: coalesced across lanes per K row
#pragma unroll
            for (int h = 0; h < 16; ++h) {
                float f = B[(size_t)(k0 + kb + h) * NDIM + coln];
                bh.s[h] = f2bf(f);
                bl.s[h] = f2bf(f - bf2f(bh.s[h]));
            }
        }

        // ---- 4 M-tiles x 3 split products ----
#pragma unroll
        for (int mt = 0; mt < 4; ++mt) {
            union { v16bf v; unsigned int u[8]; } ah, al;
            int rbase = ((lane & 15) + mt * 16) * 18;
#pragma unroll
            for (int d = 0; d < 4; ++d) {
                ah.u[d]     = AHd[rbase + ak2 + d];
                ah.u[d + 4] = AHd[rbase + ak2 + 4 + d];
                al.u[d]     = ALd[rbase + ak2 + d];
                al.u[d + 4] = ALd[rbase + ak2 + 4 + d];
            }
            acc[mt] = __builtin_amdgcn_wmma_f32_16x16x32_bf16(false, al.v, false, bh.v,
                                                              (short)0, acc[mt], false, false);
            acc[mt] = __builtin_amdgcn_wmma_f32_16x16x32_bf16(false, ah.v, false, bl.v,
                                                              (short)0, acc[mt], false, false);
            acc[mt] = __builtin_amdgcn_wmma_f32_16x16x32_bf16(false, ah.v, false, bh.v,
                                                              (short)0, acc[mt], false, false);
        }
    }

    // ---- epilogue: C layout = VGPR e -> row e (+8 for upper lane half) ----
#pragma unroll
    for (int mt = 0; mt < 4; ++mt) {
#pragma unroll
        for (int e = 0; e < 8; ++e) {
            int row = mt * 16 + (lane >> 4) * 8 + e;
            Out[row * NDIM + coln] = acc[mt][e];
        }
    }
}

// ============================================================================
// K4: Y += gamma^(2^k) * G^(2^k) @ Y @ diag(Lambda_S^(2^k))   (in place;
// each block owns a 64-column slab; Y slab cached in LDS before overwrite)
// ============================================================================
__global__ __launch_bounds__(256) void k_update(const float* __restrict__ Gp,
                                                const float* __restrict__ LS,
                                                const float* __restrict__ gp,
                                                int kidx,
                                                float* __restrict__ Y) {
    __shared__ float Gl[64][64];
    __shared__ float Yl[64][65];
    int tid = threadIdx.x;
    const float* Gk = Gp + kidx * 4096;
    for (int e = tid; e < 4096; e += 256) Gl[e >> 6][e & 63] = Gk[e];

    int j    = tid & 63;
    int colg = blockIdx.x * 64 + j;
    int ig   = (tid >> 6) * 16;
#pragma unroll
    for (int r = 0; r < 16; ++r) Yl[ig + r][j] = Y[(ig + r) * NDIM + colg];
    __syncthreads();

    float gk  = gp[0];
    float lam = LS[colg];
    for (int s = 0; s < kidx; ++s) { gk *= gk; lam *= lam; }
    float scale = gk * lam;

    float accv[16];
#pragma unroll
    for (int ii = 0; ii < 16; ++ii) accv[ii] = 0.f;
    for (int t = 0; t < 64; ++t) {
        float y = Yl[t][j];
#pragma unroll
        for (int ii = 0; ii < 16; ++ii) accv[ii] += Gl[ig + ii][t] * y;
    }
#pragma unroll
    for (int ii = 0; ii < 16; ++ii)
        Y[(ig + ii) * NDIM + colg] = Yl[ig + ii][j] + scale * accv[ii];
}

// ============================================================================
// launch
// ============================================================================
extern "C" void kernel_launch(void* const* d_in, const int* in_sizes, int n_in,
                              void* d_out, int out_size, void* d_ws, size_t ws_size,
                              hipStream_t stream) {
    (void)in_sizes; (void)n_in; (void)out_size; (void)ws_size;
    const float* X  = (const float*)d_in[0];   // (64, 4096)
    const float* F  = (const float*)d_in[1];   // (64, 64)
    const float* QS = (const float*)d_in[2];   // (4096, 4096)
    const float* LS = (const float*)d_in[3];   // (4096,)
    const float* gp = (const float*)d_in[4];   // scalar gamma
    float* Z  = (float*)d_out;                 // (64, 4096)

    float* ws = (float*)d_ws;
    float* Gp = ws;                 // 7 * 4096 floats: G^(2^k), k=0..6
    float* Y  = ws + 7 * 4096;      // 64 * 4096 floats

    k_gram<<<1, 256, 0, stream>>>(F, Gp);
    k_gpow<<<1, 256, 0, stream>>>(Gp);
    k_gemm_wmma<false><<<NDIM / 64, 128, 0, stream>>>(X, QS, Y);     // Y = X @ Q_S
    for (int k = 0; k < 7; ++k)                                      // Neumann, 128 terms
        k_update<<<NDIM / 64, 256, 0, stream>>>(Gp, LS, gp, k, Y);
    k_gemm_wmma<true><<<NDIM / 64, 128, 0, stream>>>(Y, QS, Z);      // Z = Y @ Q_S^T
}
